// BLSTM_17806934409532
// MI455X (gfx1250) — matrix operand order repaired and checked by reading, not compiled, not run on hardware
//
#include <hip/hip_runtime.h>
#include <hip/hip_bf16.h>

typedef __bf16 bf16;
typedef __attribute__((ext_vector_type(16))) __bf16 v16bf;
typedef __attribute__((ext_vector_type(8)))  float  v8f;
typedef __attribute__((ext_vector_type(4)))  unsigned int uint4v;

#define T_SEQ 512
#define BATCH 64
#define HID   512
#define DIN   1024      // layer input width (1024 for both layers: D_IN and 2H)
#define KTOT  1536      // 1024 (input) + 512 (hidden) contraction per step
#define KPAD  1544      // +16B per row: break LDS bank conflicts (3088B stride)
#define NWG_PER_DIR 32
#define WSLICE (64 * KTOT)          // bf16 elems per (l,d,wg) weight slice

union Frag { v16bf v; uint4v u[2]; };

__device__ __forceinline__ v8f splat8(float x) {
  v8f r;
#pragma unroll
  for (int i = 0; i < 8; ++i) r[i] = x;
  return r;
}
__device__ __forceinline__ float fsig(float x)  { return 1.0f / (1.0f + __expf(-x)); }
__device__ __forceinline__ float ftanh(float x) { float e = __expf(2.0f * x); return (e - 1.0f) / (e + 1.0f); }

// seq (B,T,D) fp32 -> xbuf (T,B,D) bf16
__global__ void cvt_transpose(const float* __restrict__ seq, bf16* __restrict__ xout) {
  size_t i = (size_t)blockIdx.x * 256 + threadIdx.x;
  if (i >= (size_t)BATCH * T_SEQ * DIN) return;
  int b = (int)(i / ((size_t)T_SEQ * DIN));
  size_t rem = i - (size_t)b * T_SEQ * DIN;
  int t = (int)(rem / DIN);
  int k = (int)(rem - (size_t)t * DIN);
  xout[((size_t)t * BATCH + b) * DIN + k] = (bf16)seq[i];
}

// Pre-pack all weights to bf16 in per-(layer,dir,wg) slice order:
// wbf[((l*2+d)*32 + wg)*WSLICE + gate*KTOT + k], gate = G*16 + j (G in i,f,g,o)
__global__ void cvt_weights(const float* __restrict__ w_ih, const float* __restrict__ w_hh,
                            bf16* __restrict__ wbf) {
  size_t i = (size_t)blockIdx.x * 256 + threadIdx.x;
  if (i >= (size_t)4 * NWG_PER_DIR * WSLICE) return;
  int k    = (int)(i % KTOT);
  int gate = (int)((i / KTOT) % 64);
  int wg   = (int)((i / ((size_t)KTOT * 64)) % NWG_PER_DIR);
  int ld   = (int)( i / ((size_t)KTOT * 64 * NWG_PER_DIR));   // (l*2 + d)
  int gr   = (gate >> 4) * HID + (wg << 4) + (gate & 15);
  float w  = (k < DIN) ? w_ih[((size_t)(ld * 2048 + gr)) * DIN + k]
                       : w_hh[((size_t)(ld * 2048 + gr)) * HID + (k - DIN)];
  wbf[i] = (bf16)w;
}

// One launch per layer. grid = 64 WGs (32 per direction), 128 threads (4 waves).
// WG owns 16 hidden cols -> 64 gate rows; bf16 weight slice lives in LDS (staged
// via global_load_async_to_lds_b128). Wave w owns batch rows 16w..16w+15 and all
// 4 gate tiles, so i/f/g/o for one (batch,hidden) element share a lane/VGPR slot.
// Split barrier: signal after storing h(t); wait only before the hidden-K region
// of step t+1, hiding agent-scope barrier latency under the input-projection GEMM.
__global__ __launch_bounds__(128, 1) void lstm_layer(
    const bf16* __restrict__ wbf,
    const float* __restrict__ b_ih, const float* __restrict__ b_hh,
    const bf16*  __restrict__ xin,  bf16* __restrict__ hbuf,
    float* __restrict__ dout, int layer, unsigned int* __restrict__ bar) {

  __shared__ bf16 wsl[64 * KPAD];   // ~193 KB of the WGP's 320 KB LDS

  const int tid = threadIdx.x;
  const int dir = blockIdx.x >> 5;
  const int wg  = blockIdx.x & 31;
  const int j0  = wg << 4;
  const int ldBase = (layer * 2 + dir) * 2048;

  // ---- stage bf16 weight slice into LDS with async DMA (ASYNCcnt path) ----
  {
    const bf16* slice = wbf + ((size_t)((layer * 2 + dir) * NWG_PER_DIR + wg)) * WSLICE;
    // 16B chunks: 96 per gate row (first 768 of the KTOT bf16), 6144 total, 48 per thread
    for (int c = tid; c < 64 * 96; c += 128) {
      int gate = c / 96;
      int kc   = (c - gate * 96) * 8;                    // bf16 element offset in row
      unsigned ldsb = (unsigned)(unsigned long long)(uintptr_t)(&wsl[gate * KPAD + kc]);
      unsigned long long ga = (unsigned long long)(uintptr_t)(slice + (size_t)gate * KTOT + kc);
      asm volatile("global_load_async_to_lds_b128 %0, %1, off"
                   :: "v"(ldsb), "v"(ga) : "memory");
    }
    asm volatile("s_wait_asynccnt 0x0" ::: "memory");
  }
  __syncthreads();

  const int wave = tid >> 5, lane = tid & 31;
  const int n = lane & 15, laneHi = lane >> 4;
  const int kbA = laneHi ? 8 : 0;    // A frag: K interleave 0-7 / 16-23 (+8 for hi lanes)
  const int kbB = laneHi ? 8 : 0;    // B frag: K interleave 0-7 / 16-23 (+8 for hi lanes)
  const int mrow = (wave << 4) + n;  // batch row this lane loads for A
  const int dirOff = dir ? HID : 0;

  float bias[4];
#pragma unroll
  for (int G = 0; G < 4; ++G) {
    int gr = ldBase + G * HID + j0 + n;
    bias[G] = b_ih[gr] + b_hh[gr];
  }

  v8f c = splat8(0.0f);

  for (int step = 0; step < T_SEQ; ++step) {
    const int tt = dir ? (T_SEQ - 1 - step) : step;

    v8f acc[4];
#pragma unroll
    for (int G = 0; G < 4; ++G) acc[G] = splat8(bias[G]);

    if (step + 1 < T_SEQ) {   // pull next timestep's activations toward the WGP
      int tn = dir ? tt - 1 : tt + 1;
      __builtin_prefetch(xin + ((size_t)tn * BATCH + mrow) * DIN, 0, 1);
    }

    // ---- input projection: K = 0..1023 from xin[t] (no cross-WG dependency) ----
    const bf16* arow = xin + ((size_t)tt * BATCH + mrow) * DIN;
    for (int kk = 0; kk < DIN / 32; ++kk) {
      Frag a;
      const bf16* ap = arow + kk * 32 + kbA;
      a.u[0] = *(const uint4v*)(ap);
      a.u[1] = *(const uint4v*)(ap + 16);
#pragma unroll
      for (int G = 0; G < 4; ++G) {
        Frag bf_;
        const bf16* bp = &wsl[(G * 16 + n) * KPAD + kk * 32 + kbB];
        bf_.u[0] = *(const uint4v*)(bp);
        bf_.u[1] = *(const uint4v*)(bp + 16);
        acc[G] = __builtin_amdgcn_wmma_f32_16x16x32_bf16(
            false, a.v, false, bf_.v, (short)0, acc[G], false, false);
      }
    }

    // ---- wait for h(t-1), then recurrence: K = 1024..1535 ----
    if (step > 0) {
      if (tid == 0) {
        unsigned target = (unsigned)NWG_PER_DIR * (unsigned)step;  // all WGs done step-1
        while (__hip_atomic_load(&bar[dir], __ATOMIC_ACQUIRE, __HIP_MEMORY_SCOPE_AGENT) < target)
          __builtin_amdgcn_s_sleep(1);
      }
      __syncthreads();
      __threadfence();   // acquire h(t-1) stores at agent scope for all waves

      const int tp = dir ? tt + 1 : tt - 1;
      const bf16* hrow = hbuf + ((size_t)tp * BATCH + mrow) * DIN + dirOff;
      for (int kk = 0; kk < HID / 32; ++kk) {
        Frag a;
        const bf16* ap = hrow + kk * 32 + kbA;
        a.u[0] = *(const uint4v*)(ap);
        a.u[1] = *(const uint4v*)(ap + 16);
#pragma unroll
        for (int G = 0; G < 4; ++G) {
          Frag bf_;
          const bf16* bp = &wsl[(G * 16 + n) * KPAD + DIN + kk * 32 + kbB];
          bf_.u[0] = *(const uint4v*)(bp);
          bf_.u[1] = *(const uint4v*)(bp + 16);
          acc[G] = __builtin_amdgcn_wmma_f32_16x16x32_bf16(
              false, a.v, false, bf_.v, (short)0, acc[G], false, false);
        }
      }
    }

    // ---- gate nonlinearity + cell/hidden update (pure per-lane VALU) ----
#pragma unroll
    for (int r = 0; r < 8; ++r) {
      float iv = fsig(acc[0][r]);
      float fv = fsig(acc[1][r]);
      float gv = ftanh(acc[2][r]);
      float ov = fsig(acc[3][r]);
      float cn = fv * c[r] + iv * gv;
      c[r] = cn;
      float hv = ov * ftanh(cn);
      int b = (wave << 4) + r + (laneHi << 3);          // C/D layout: M = r + 8*laneHi
      hbuf[((size_t)tt * BATCH + b) * DIN + dirOff + j0 + n] = (bf16)hv;
      if (dout) dout[((size_t)b * T_SEQ + tt) * (2 * HID) + dirOff + j0 + n] = hv;
    }

    // ---- signal h(t) complete (no trailing wait: consumers wait mid-step) ----
    __threadfence();
    __syncthreads();
    if (tid == 0)
      __hip_atomic_fetch_add(&bar[dir], 1u, __ATOMIC_RELEASE, __HIP_MEMORY_SCOPE_AGENT);
  }
}

extern "C" void kernel_launch(void* const* d_in, const int* in_sizes, int n_in,
                              void* d_out, int out_size, void* d_ws, size_t ws_size,
                              hipStream_t stream) {
  const float* seq  = (const float*)d_in[0];
  // d_in[1] = mask (unused by the reference)
  const float* w_ih = (const float*)d_in[2];
  const float* w_hh = (const float*)d_in[3];
  const float* b_ih = (const float*)d_in[4];
  const float* b_hh = (const float*)d_in[5];
  float* out = (float*)d_out;

  const size_t elems = (size_t)T_SEQ * BATCH * DIN;     // 33.5M activations
  const size_t welems = (size_t)4 * NWG_PER_DIR * WSLICE; // 12.6M bf16 weights
  bf16* xbuf0 = (bf16*)d_ws;                            // layer0 input / layer1 h-scratch
  bf16* xbuf1 = xbuf0 + elems;                          // layer0 output = layer1 input
  bf16* wbf   = xbuf1 + elems;                          // packed bf16 weights
  unsigned int* bar = (unsigned int*)(wbf + welems);

  cvt_transpose<<<(unsigned)((elems + 255) / 256), 256, 0, stream>>>(seq, xbuf0);
  cvt_weights<<<(unsigned)((welems + 255) / 256), 256, 0, stream>>>(w_ih, w_hh, wbf);

  hipMemsetAsync(bar, 0, 2 * sizeof(unsigned int), stream);
  lstm_layer<<<64, 128, 0, stream>>>(wbf, b_ih, b_hh, xbuf0, xbuf1, nullptr, 0, bar);

  hipMemsetAsync(bar, 0, 2 * sizeof(unsigned int), stream);
  lstm_layer<<<64, 128, 0, stream>>>(wbf, b_ih, b_hh, xbuf1, xbuf0, out, 1, bar);
}
